// BahdanauAttention_63393717289501
// MI455X (gfx1250) — matrix operand order repaired and checked by reading, not compiled, not run on hardware
//
#include <hip/hip_runtime.h>
#include <hip/hip_bf16.h>

#define BATCH 32
#define SLEN  1024
#define DH    512
#define EH2   1024
#define WROW  1536
#define NEGV  (-10000000000.0f)

typedef __attribute__((ext_vector_type(16))) __bf16 v16bf;
typedef __attribute__((ext_vector_type(8)))  float  v8f;

union Frag16 {
    uint4 u[2];
    v16bf v;
};

__device__ __forceinline__ unsigned int pack_bf16(float a, float b) {
    unsigned int ua = __float_as_uint(a), ub = __float_as_uint(b);
    ua = (ua + 0x7FFFu + ((ua >> 16) & 1u)) >> 16;
    ub = (ub + 0x7FFFu + ((ub >> 16) & 1u)) >> 16;
    return (ub << 16) | (ua & 0xFFFFu);
}

// Branch-free tanh: tanh(x) = 1 - 2/(exp(2x)+1).  ~1e-6 rel error, no exec
// divergence.
__device__ __forceinline__ float fast_tanh(float x) {
    x = fminf(10.0f, fmaxf(-10.0f, x));
    const float e = __expf(2.0f * x);
    return 1.0f - __fdividef(2.0f, e + 1.0f);
}

// B fragments for 4 N-tiles at K-offset kbase (8 x global_load_b128, one clause).
__device__ __forceinline__ void load_bfrag(Frag16* dst, const unsigned short* bp0,
                                           int kbase) {
#pragma unroll
    for (int t = 0; t < 4; ++t) {
        const unsigned short* bp = bp0 + (size_t)t * 16 * EH2 + kbase;
        dst[t].u[0] = *(const uint4*)(bp);
        dst[t].u[1] = *(const uint4*)(bp + 8);
    }
}

// A fragment from LDS at K-offset kbase (2 x ds_load_b128).
__device__ __forceinline__ v16bf load_afrag(const unsigned short* aLane, int kbase) {
    Frag16 a;
    a.u[0] = *(const uint4*)(aLane + kbase);
    a.u[1] = *(const uint4*)(aLane + kbase + 8);
    return a.v;
}

// ---------------------------------------------------------------------------
// K1: h_proj[b,d] = b[d] + sum_k hidden[b,k] * W[d, k]   (Wh = W[:, :512])
// ---------------------------------------------------------------------------
__global__ __launch_bounds__(256) void hproj_kernel(const float* __restrict__ hidden,
                                                    const float* __restrict__ W,
                                                    const float* __restrict__ bvec,
                                                    float* __restrict__ hp) {
    __shared__ float hsh[DH];
    const int b = blockIdx.x >> 1;
    const int d = ((blockIdx.x & 1) << 8) + threadIdx.x;

    hsh[threadIdx.x]       = hidden[b * DH + threadIdx.x];
    hsh[threadIdx.x + 256] = hidden[b * DH + threadIdx.x + 256];
    __syncthreads();

    const float4* w = (const float4*)(W + (size_t)d * WROW);
    float s = bvec[d];
#pragma unroll 4
    for (int k4 = 0; k4 < 128; ++k4) {
        const float4 wv = w[k4];
        s += hsh[k4 * 4 + 0] * wv.x + hsh[k4 * 4 + 1] * wv.y +
             hsh[k4 * 4 + 2] * wv.z + hsh[k4 * 4 + 3] * wv.w;
    }
    hp[b * DH + d] = s;
}

// ---------------------------------------------------------------------------
// K2: We (= W[:, 512:1536]) f32 -> bf16, row-major (512 x 1024).
// ---------------------------------------------------------------------------
__global__ __launch_bounds__(256) void wecvt_kernel(const float* __restrict__ W,
                                                    unsigned int* __restrict__ web) {
    const int idx = blockIdx.x * 256 + threadIdx.x;  // 0 .. 262143
    const int d   = idx >> 9;
    const int ep  = idx & 511;
    const float* src = W + (size_t)d * WROW + DH + ep * 2;
    web[(size_t)d * 512 + ep] = pack_bf16(src[0], src[1]);
}

// ---------------------------------------------------------------------------
// K3: main fused kernel. One workgroup per (b, 16-row s-tile) = 2048 WGs.
// Software-pipelined: B fragments for K-step kb+1 are in flight while the
// 4 WMMAs of K-step kb execute (partial s_wait_loadcnt instead of full drain).
// ---------------------------------------------------------------------------
__global__ __launch_bounds__(256) void attn_main_kernel(const float* __restrict__ enc,
                                                        const float* __restrict__ hp,
                                                        const unsigned short* __restrict__ weB,
                                                        const float* __restrict__ vvec,
                                                        const int* __restrict__ mask,
                                                        float* __restrict__ scores) {
    __shared__ __align__(16) unsigned short Abf[16 * 1032];
    __shared__ float wpart[8][16];

    const int tid   = threadIdx.x;
    const int wave  = tid >> 5;
    const int lane  = tid & 31;
    const int laneM = lane & 15;
    const int hi    = lane >> 4;

    const int b  = blockIdx.x >> 6;
    const int st = blockIdx.x & 63;
    const float* encBase = enc + ((size_t)(b * SLEN + st * 16)) * EH2;

    // --- stage A: 16 x 1024 f32 -> bf16 in LDS (b128 stores, coalesced) ---
    {
        const int rhalf = tid >> 7;
        const int kt    = (tid & 127) * 8;
#pragma unroll
        for (int i = 0; i < 8; ++i) {
            const int row = i * 2 + rhalf;
            const float4 f0 = *(const float4*)(encBase + (size_t)row * EH2 + kt);
            const float4 f1 = *(const float4*)(encBase + (size_t)row * EH2 + kt + 4);
            uint4 p;
            p.x = pack_bf16(f0.x, f0.y);
            p.y = pack_bf16(f0.z, f0.w);
            p.z = pack_bf16(f1.x, f1.y);
            p.w = pack_bf16(f1.z, f1.w);
            *(uint4*)(&Abf[row * 1032 + kt]) = p;
        }
    }
    __syncthreads();

    // --- GEMM: 4 N-tiles per wave, 32 K-steps, double-buffered B ---
    const int nbase0 = wave * 64;
    v8f acc[4];
#pragma unroll
    for (int t = 0; t < 4; ++t)
#pragma unroll
        for (int r = 0; r < 8; ++r) acc[t][r] = 0.0f;

    const unsigned short* bp0   = weB + (size_t)(nbase0 + laneM) * EH2 + hi * 16;
    const unsigned short* aLane = &Abf[laneM * 1032 + hi * 16];

    Frag16 b0[4], b1[4];
    load_bfrag(b0, bp0, 0);

    for (int kb = 0; kb < 32; kb += 2) {
        // A fragments for both half-steps early (ds latency hidden by wmmas)
        const v16bf a0 = load_afrag(aLane, kb * 32);
        const v16bf a1 = load_afrag(aLane, kb * 32 + 32);

        // prefetch B for kb+1 while computing on b0
        load_bfrag(b1, bp0, (kb + 1) * 32);
#pragma unroll
        for (int t = 0; t < 4; ++t)
            acc[t] = __builtin_amdgcn_wmma_f32_16x16x32_bf16(
                false, a0, false, b0[t].v, (short)0, acc[t], false, false);

        // prefetch B for kb+2 (clamped scalar offset keeps the body branch-free)
        const int kn = (kb + 2 < 32) ? (kb + 2) * 32 : 0;
        load_bfrag(b0, bp0, kn);
#pragma unroll
        for (int t = 0; t < 4; ++t)
            acc[t] = __builtin_amdgcn_wmma_f32_16x16x32_bf16(
                false, a1, false, b1[t].v, (short)0, acc[t], false, false);
    }

    // --- fused epilogue: tanh + dot(v), branch-free ---
    const float* hpb = hp + b * DH;
    float part[8];
#pragma unroll
    for (int r = 0; r < 8; ++r) part[r] = 0.0f;
#pragma unroll
    for (int t = 0; t < 4; ++t) {
        const int d   = nbase0 + t * 16 + laneM;
        const float vd  = vvec[d];
        const float hpd = hpb[d];
#pragma unroll
        for (int r = 0; r < 8; ++r)
            part[r] += vd * fast_tanh(hpd + acc[t][r]);
    }
#pragma unroll
    for (int r = 0; r < 8; ++r) {
        float x = part[r];
        x += __shfl_xor(x, 1, 16);
        x += __shfl_xor(x, 2, 16);
        x += __shfl_xor(x, 4, 16);
        x += __shfl_xor(x, 8, 16);
        if (laneM == 0) wpart[wave][r + 8 * hi] = x;   // m = r + 8*hi
    }
    __syncthreads();

    if (tid < 16) {
        float s = 0.0f;
#pragma unroll
        for (int w = 0; w < 8; ++w) s += wpart[w][tid];
        const int sIdx = st * 16 + tid;
        const int mk = mask[b * SLEN + sIdx];
        scores[b * SLEN + sIdx] = (mk == 0) ? NEGV : s;
    }
}

// ---------------------------------------------------------------------------
// K4: softmax over S=1024 per batch row.
// ---------------------------------------------------------------------------
__global__ __launch_bounds__(256) void softmax_kernel(const float* __restrict__ scores,
                                                      float* __restrict__ out) {
    __shared__ float red[8];
    const int b    = blockIdx.x;
    const int tid  = threadIdx.x;
    const int wave = tid >> 5;
    const int lane = tid & 31;
    const float* row = scores + b * SLEN;

    float x[4];
    float mx = -3.4e38f;
#pragma unroll
    for (int i = 0; i < 4; ++i) {
        x[i] = row[tid + i * 256];
        mx = fmaxf(mx, x[i]);
    }
#pragma unroll
    for (int m = 1; m < 32; m <<= 1) mx = fmaxf(mx, __shfl_xor(mx, m, 32));
    if (lane == 0) red[wave] = mx;
    __syncthreads();
    float gmx = red[0];
#pragma unroll
    for (int w = 1; w < 8; ++w) gmx = fmaxf(gmx, red[w]);
    __syncthreads();

    float sum = 0.0f;
#pragma unroll
    for (int i = 0; i < 4; ++i) {
        x[i] = __expf(x[i] - gmx);
        sum += x[i];
    }
#pragma unroll
    for (int m = 1; m < 32; m <<= 1) sum += __shfl_xor(sum, m, 32);
    if (lane == 0) red[wave] = sum;
    __syncthreads();
    float gs = 0.0f;
#pragma unroll
    for (int w = 0; w < 8; ++w) gs += red[w];
    const float inv = 1.0f / gs;
#pragma unroll
    for (int i = 0; i < 4; ++i) out[b * SLEN + tid + i * 256] = x[i] * inv;
}

// ---------------------------------------------------------------------------
extern "C" void kernel_launch(void* const* d_in, const int* in_sizes, int n_in,
                              void* d_out, int out_size, void* d_ws, size_t ws_size,
                              hipStream_t stream) {
    const float* hidden = (const float*)d_in[0];  // (32, 512)
    const float* enc    = (const float*)d_in[1];  // (32, 1024, 1024)
    const int*   mask   = (const int*)d_in[2];    // (32, 1024)
    const float* W      = (const float*)d_in[3];  // (512, 1536)
    const float* bvec   = (const float*)d_in[4];  // (512,)
    const float* vvec   = (const float*)d_in[5];  // (512,)
    float* out = (float*)d_out;                   // (32, 1024)

    char* ws = (char*)d_ws;
    float*        hp     = (float*)ws;                            // 64 KB
    unsigned int* web    = (unsigned int*)(ws + 65536);           // 1 MB bf16 We
    float*        scores = (float*)(ws + 65536 + 1048576);        // 128 KB

    hproj_kernel<<<64, 256, 0, stream>>>(hidden, W, bvec, hp);
    wecvt_kernel<<<1024, 256, 0, stream>>>(W, web);
    attn_main_kernel<<<BATCH * (SLEN / 16), 256, 0, stream>>>(
        enc, hp, (const unsigned short*)web, vvec, mask, scores);
    softmax_kernel<<<32, 256, 0, stream>>>(scores, out);
}
